// NAM_5222680232186
// MI455X (gfx1250) — matrix operand order repaired and checked by reading, not compiled, not run on hardware
//
#include <hip/hip_runtime.h>

typedef __attribute__((ext_vector_type(16))) _Float16 v16h;
typedef __attribute__((ext_vector_type(8)))  _Float16 v8h;
typedef __attribute__((ext_vector_type(8)))  float    v8f;
typedef __attribute__((ext_vector_type(4)))  float    v4f;
typedef unsigned __attribute__((ext_vector_type(2))) uint2v;

#define BATCH 2048
#define DIM   256

// ---------------- LDS layout (bytes) ----------------
constexpr int OFF_W1 = 0;                    // 64x32   f16
constexpr int OFF_W2 = OFF_W1 + 64 * 32 * 2; // 128x64  f16
constexpr int OFF_W3 = OFF_W2 + 128 * 64 * 2;// 64x128  f16
constexpr int OFF_W4 = OFF_W3 + 64 * 128 * 2;// 32x64   f16
constexpr int OFF_W5 = OFF_W4 + 32 * 64 * 2; // 16x32   f16
constexpr int OFF_F  = OFF_W5 + 16 * 32 * 2; // float region
constexpr int F_W0 = 0,  F_W6 = 32, F_B0 = 48, F_B1 = 80, F_B2 = 144;
constexpr int F_B3 = 272, F_B4 = 336, F_B5 = 368, F_B6 = 384;
constexpr int NFLOATS = 385;
constexpr int LDS_BYTES = ((OFF_F + NFLOATS * 4 + 255) / 256) * 256; // ~43.5 KB

#define SWAPX16_IMM 0x401F  // ds_swizzle group-of-32: xor=0x10, or=0, and=0x1f
#define ANTI_HOIST() asm volatile("" ::: "memory")

// f32 pair -> packed f16 + ReLU: exactly 2 VALU ops.
__device__ __forceinline__ int cvt_relu(float a, float b) {
  int r;
  asm("v_cvt_pk_f16_f32 %0, %1, %2\n\t"
      "v_pk_max_num_f16 %0, %0, 0"
      : "=v"(r) : "v"(a), "v"(b));
  return r;
}

__device__ __forceinline__ float relu_f32(float v) {
  float r;
  asm("v_max_num_f32 %0, %1, 0" : "=v"(r) : "v"(v));
  return r;
}

// Swap lane-halves between two dwords (D split -> B split conversion):
//   ret0 = {t0.lanes[0:15],  t1.lanes[0:15]}
//   ret1 = {t0.lanes[16:31], t1.lanes[16:31]}
__device__ __forceinline__ uint2v half_swap(int t0, int t1) {
#if __has_builtin(__builtin_amdgcn_permlane16_swap)
  return __builtin_amdgcn_permlane16_swap((unsigned)t0, (unsigned)t1,
                                          false, false);
#else
  const int hh = (int)(threadIdx.x & 31) >> 4;
  int s = hh ? t0 : t1;
  int p = __builtin_amdgcn_ds_swizzle(s, SWAPX16_IMM);
  uint2v res;
  res[0] = hh ? p : t0;
  res[1] = hh ? t1 : p;
  return res;
#endif
}

// Two f32 D tiles (feature pair) -> one B-layout v16h (relu + f16 quantize).
__device__ __forceinline__ void pack_swap(const v8f& c0, const v8f& c1,
                                          v16h& out) {
  union { unsigned u[8]; v16h v; } U;
#pragma unroll
  for (int i = 0; i < 4; ++i) {
    int t0 = cvt_relu(c0[2 * i], c0[2 * i + 1]);
    int t1 = cvt_relu(c1[2 * i], c1[2 * i + 1]);
    uint2v sw = half_swap(t0, t1);
    U.u[i]     = sw[0];  // B elems 0..7  (features 16hh+0..7)
    U.u[4 + i] = sw[1];  // B elems 8..15 (features 16hh+8..15)
  }
  out = U.v;
}

// One M tile (16 output features) for TWO batch streams: A loaded once,
// two independent WMMAs per K chunk (hides XDL latency, halves ds/wmma).
template <int KS, int FI>
__device__ __forceinline__ void dtile2(const _Float16* __restrict__ wT,
                                       const float* __restrict__ bias,
                                       const v16h* BinA, const v16h* BinB,
                                       int mt, int r, int hh,
                                       v8f& ca, v8f& cb) {
  const float* bp = bias + mt * 16 + 8 * hh;
  v4f bl = *(const v4f*)bp;
  v4f bh = *(const v4f*)(bp + 4);
  v8f c = {bl[0], bl[1], bl[2], bl[3], bh[0], bh[1], bh[2], bh[3]};
  ca = c;
  cb = c;
#pragma unroll
  for (int k = 0; k < KS; ++k) {
    const _Float16* ar = wT + (mt * 16 + r) * FI + 32 * k + 16 * hh;
    v8h lo = *(const v8h*)ar;
    v8h hi = *(const v8h*)(ar + 8);
    v16h a;
#pragma unroll
    for (int j = 0; j < 8; ++j) { a[j] = lo[j]; a[8 + j] = hi[j]; }
    ca = __builtin_amdgcn_wmma_f32_16x16x32_f16(false, a, false, BinA[k],
                                                (short)0, ca, false, false);
    cb = __builtin_amdgcn_wmma_f32_16x16x32_f16(false, a, false, BinB[k],
                                                (short)0, cb, false, false);
  }
}

// Full layer for two batch streams.
template <int FI, int FO>
__device__ __forceinline__ void layer2_wmma(const _Float16* __restrict__ wT,
                                            const float* __restrict__ bias,
                                            const v16h (&BinA)[FI / 32],
                                            const v16h (&BinB)[FI / 32],
                                            v16h (&BoutA)[FO / 32],
                                            v16h (&BoutB)[FO / 32],
                                            int r, int hh) {
  constexpr int KS = FI / 32;
#pragma unroll
  for (int mp = 0; mp < FO / 32; ++mp) {
    v8f c0a, c0b, c1a, c1b;
    dtile2<KS, FI>(wT, bias, BinA, BinB, 2 * mp,     r, hh, c0a, c0b);
    dtile2<KS, FI>(wT, bias, BinA, BinB, 2 * mp + 1, r, hh, c1a, c1b);
    pack_swap(c0a, c1a, BoutA[mp]);
    pack_swap(c0b, c1b, BoutB[mp]);
  }
}

extern "C" __global__ void __launch_bounds__(256)
mlp_channel_kernel(const float* __restrict__ x,
                   const float* __restrict__ W0, const float* __restrict__ b0,
                   const float* __restrict__ W1, const float* __restrict__ b1,
                   const float* __restrict__ W2, const float* __restrict__ b2,
                   const float* __restrict__ W3, const float* __restrict__ b3,
                   const float* __restrict__ W4, const float* __restrict__ b4,
                   const float* __restrict__ W5, const float* __restrict__ b5,
                   const float* __restrict__ W6, const float* __restrict__ b6,
                   float* __restrict__ ws) {
  extern __shared__ char smem[];
  const int d    = blockIdx.x;
  const int half = blockIdx.y;
  const int tid  = threadIdx.x;

  _Float16* w1 = (_Float16*)(smem + OFF_W1);
  _Float16* w2 = (_Float16*)(smem + OFF_W2);
  _Float16* w3 = (_Float16*)(smem + OFF_W3);
  _Float16* w4 = (_Float16*)(smem + OFF_W4);
  _Float16* w5 = (_Float16*)(smem + OFF_W5);
  float*    fr = (float*)(smem + OFF_F);

  auto loadT = [&](const float* G, _Float16* dst, int fi, int fo) {
    const int n = fi * fo;
    const float* g = G + d * n;
    for (int idx = tid; idx < n; idx += 256) {
      const int i = idx / fo, o = idx % fo;
      dst[o * fi + i] = (_Float16)g[idx];
    }
  };
  loadT(W1, w1, 32, 64);
  loadT(W2, w2, 64, 128);
  loadT(W3, w3, 128, 64);
  loadT(W4, w4, 64, 32);
  loadT(W5, w5, 32, 16);
  for (int i = tid; i < 32; i += 256)  fr[F_W0 + i] = W0[d * 32 + i];
  for (int i = tid; i < 16; i += 256)  fr[F_W6 + i] = W6[d * 16 + i];
  for (int i = tid; i < 32; i += 256)  fr[F_B0 + i] = b0[d * 32 + i];
  for (int i = tid; i < 64; i += 256)  fr[F_B1 + i] = b1[d * 64 + i];
  for (int i = tid; i < 128; i += 256) fr[F_B2 + i] = b2[d * 128 + i];
  for (int i = tid; i < 64; i += 256)  fr[F_B3 + i] = b3[d * 64 + i];
  for (int i = tid; i < 32; i += 256)  fr[F_B4 + i] = b4[d * 32 + i];
  for (int i = tid; i < 16; i += 256)  fr[F_B5 + i] = b5[d * 16 + i];
  if (tid == 0) fr[F_B6] = b6[d];
  __syncthreads();

  const int wave = tid >> 5;  // wave32
  const int lane = tid & 31;
  const int r  = lane & 15;   // batch column within tile
  const int hh = lane >> 4;   // lane half

  const int t0i = half * 64;  // 64 tiles per batch half; 8 per wave, paired
#pragma clang loop unroll(disable)
  for (int it = 0; it < 4; ++it) {
    const int tA    = t0i + wave + it * 16;
    const int row0a = tA * 16;
    const int row0b = row0a + 128;  // tile tA+8
    ANTI_HOIST();  // keep weight/bias LDS loads inside the iteration

    // ---- Layer 0 (1 -> 32): build B operands in registers ----
    v16h B0a[1], B0b[1];
    {
      const float xa = x[(row0a + r) * DIM + d];
      const float xb = x[(row0b + r) * DIM + d];
      union { unsigned u[8]; v16h v; } Ua, Ub;
#pragma unroll
      for (int i = 0; i < 8; ++i) {
        const int f = 16 * hh + 2 * i;
        const float w0l = fr[F_W0 + f], w0h = fr[F_W0 + f + 1];
        const float bl0 = fr[F_B0 + f], bh0 = fr[F_B0 + f + 1];
        Ua.u[i] = cvt_relu(xa * w0l + bl0, xa * w0h + bh0);
        Ub.u[i] = cvt_relu(xb * w0l + bl0, xb * w0h + bh0);
      }
      B0a[0] = Ua.v;
      B0b[0] = Ub.v;
    }

    // ---- WMMA layers, activations resident in registers ----
    v16h B1a[2], B1b[2], B2a[4], B2b[4], B3a[2], B3b[2], B4a[1], B4b[1];
    layer2_wmma<32, 64>(w1, fr + F_B1, B0a, B0b, B1a, B1b, r, hh);
    layer2_wmma<64, 128>(w2, fr + F_B2, B1a, B1b, B2a, B2b, r, hh);
    layer2_wmma<128, 64>(w3, fr + F_B3, B2a, B2b, B3a, B3b, r, hh);
    layer2_wmma<64, 32>(w4, fr + F_B4, B3a, B3b, B4a, B4b, r, hh);

    // ---- Layer 5 (32 -> 16): one WMMA per stream, keep f32 D ----
    v8f D5a, D5b;
    dtile2<1, 32>(w5, fr + F_B5, B4a, B4b, 0, r, hh, D5a, D5b);

    // ---- Layer 6 (16 -> 1): per-lane dot + half-swap combine ----
    float pa = 0.f, pb = 0.f;
#pragma unroll
    for (int e = 0; e < 8; ++e) {
      const float w6v = fr[F_W6 + e + 8 * hh];
      pa += relu_f32(D5a[e]) * w6v;
      pb += relu_f32(D5b[e]) * w6v;
    }
    const float oa = __int_as_float(
        __builtin_amdgcn_ds_swizzle(__float_as_int(pa), SWAPX16_IMM));
    const float ob = __int_as_float(
        __builtin_amdgcn_ds_swizzle(__float_as_int(pb), SWAPX16_IMM));
    if (hh == 0) {
      const float bb = fr[F_B6];
      ws[d * BATCH + row0a + r] = pa + oa + bb;
      ws[d * BATCH + row0b + r] = pb + ob + bb;
    }
  }
}

// Deterministic fixed-order reduction over channels: out[b] = sum_d ws[d][b]
extern "C" __global__ void __launch_bounds__(256)
reduce_channels(const float* __restrict__ ws, float* __restrict__ out) {
  const int b = blockIdx.x * 256 + threadIdx.x;
  if (b < BATCH) {
    float s = 0.f;
    for (int d = 0; d < DIM; ++d) s += ws[d * BATCH + b];
    out[b] = s;
  }
}

extern "C" void kernel_launch(void* const* d_in, const int* in_sizes, int n_in,
                              void* d_out, int out_size, void* d_ws, size_t ws_size,
                              hipStream_t stream) {
  (void)in_sizes; (void)n_in; (void)out_size; (void)ws_size;
  const float* x  = (const float*)d_in[0];
  const float* W0 = (const float*)d_in[1];  const float* b0 = (const float*)d_in[2];
  const float* W1 = (const float*)d_in[3];  const float* b1 = (const float*)d_in[4];
  const float* W2 = (const float*)d_in[5];  const float* b2 = (const float*)d_in[6];
  const float* W3 = (const float*)d_in[7];  const float* b3 = (const float*)d_in[8];
  const float* W4 = (const float*)d_in[9];  const float* b4 = (const float*)d_in[10];
  const float* W5 = (const float*)d_in[11]; const float* b5 = (const float*)d_in[12];
  const float* W6 = (const float*)d_in[13]; const float* b6 = (const float*)d_in[14];
  float* ws = (float*)d_ws;  // needs DIM*BATCH*4 = 2 MB

  dim3 grid(DIM, 2);
  mlp_channel_kernel<<<grid, 256, LDS_BYTES, stream>>>(
      x, W0, b0, W1, b1, W2, b2, W3, b3, W4, b4, W5, b5, W6, b6, ws);
  reduce_channels<<<BATCH / 256, 256, 0, stream>>>(ws, (float*)d_out);
}
